// ClassificationHead_83373905150078
// MI455X (gfx1250) — matrix operand order repaired and checked by reading, not compiled, not run on hardware
//
#include <hip/hip_runtime.h>
#include <math.h>

#ifndef __has_builtin
#define __has_builtin(x) 0
#endif
#if __has_builtin(__builtin_amdgcn_tensor_load_to_lds)
#define USE_TDM 1
#else
#define USE_TDM 0
#endif

typedef __attribute__((ext_vector_type(16))) __bf16        bf16x16;
typedef __attribute__((ext_vector_type(8)))  float         f32x8;
typedef __attribute__((ext_vector_type(4)))  unsigned int  u32x4;
typedef __attribute__((ext_vector_type(8)))  int           i32x8;
typedef __attribute__((ext_vector_type(4)))  int           i32x4;

#define BATCH       65536
#define K_DIM       1024
#define N_DIM       256
#define ROWS        64
#define KSTEP       32
#define NCHUNKS     (K_DIM / KSTEP)          // 32
#define THREADS     256
#define XN_STRIDE_E 1032                     // bf16 elems/row (1024 + 8 pad)
#define XN_BYTES    (ROWS * XN_STRIDE_E * 2) // 132096
#define WB_ROW_B    80                       // 64B data + 16B pad per W row
#define WB_BYTES    (N_DIM * WB_ROW_B)       // 20480
#define H_STRIDE    260
#define SMEM_BYTES  (XN_BYTES + 2 * WB_BYTES)

__device__ __forceinline__ unsigned short f2bf(float f) {
  unsigned u = __float_as_uint(f);
  u += 0x7FFFu + ((u >> 16) & 1u);           // round-to-nearest-even
  return (unsigned short)(u >> 16);
}

union Frag { bf16x16 v; u32x4 q[2]; };

// ---- Tensor Data Mover: load one 32x256 bf16 W-chunk (column slice of W1^T)
// into LDS with 16B pad per 64B row (-> 80B LDS stride). D# per CDNA5 ISA.
// This toolchain's builtin arity is 6: (u32x4, i32x8, i32x4, i32x4, i32x8, i32).
__device__ __forceinline__ void tdm_issue(const unsigned short* gsrc, unsigned lds_off) {
#if USE_TDM
  unsigned long long ga = (unsigned long long)(uintptr_t)gsrc;
  u32x4 g0;
  g0[0] = 1u;                                       // count=1, user mode
  g0[1] = lds_off;                                  // lds_addr (bytes)
  g0[2] = (unsigned)ga;                             // global_addr[31:0]
  g0[3] = (unsigned)((ga >> 32) & 0x01FFFFFFu) | (2u << 30); // addr[56:32] | type=2
  i32x8 g1;
  g1[0] = (int)((1u << 16)      // data_size = 2B
              | (1u << 20)      // pad_enable
              | (3u << 22)      // pad_interval: 16 dwords (=64B row)
              | (3u << 25));    // pad_amount: 4 dwords (=16B)
  g1[1] = (int)(1024u << 16);   // tensor_dim0[15:0] = 1024
  g1[2] = (int)(256u << 16);    // tensor_dim0[31:16]=0 | tensor_dim1[15:0]=256
  g1[3] = (int)(32u << 16);     // tensor_dim1[31:16]=0 | tile_dim0=32
  g1[4] = 256;                  // tile_dim1=256 | tile_dim2=0
  g1[5] = 1024;                 // tensor_dim0_stride = 1024 elems
  g1[6] = 0;
  g1[7] = 0;
  i32x4 z4; z4[0] = 0; z4[1] = 0; z4[2] = 0; z4[3] = 0;
  i32x8 z8;
#pragma unroll
  for (int i = 0; i < 8; ++i) z8[i] = 0;
  __builtin_amdgcn_tensor_load_to_lds(g0, g1, z4, z4, z8, 0);
#else
  (void)gsrc; (void)lds_off;
#endif
}

__global__ void prep_w1_kernel(const float* __restrict__ W1,
                               unsigned short* __restrict__ W1t) {
  const int idx = blockIdx.x * 256 + threadIdx.x;  // coalesced read of W1[k][n]
  const int k = idx >> 8;
  const int n = idx & 255;
  W1t[n * K_DIM + k] = f2bf(W1[idx]);
}

__global__ __launch_bounds__(THREADS, 1)
void head_main_kernel(const float* __restrict__ X, const float* __restrict__ lnw,
                      const float* __restrict__ lnb,
                      const unsigned short* __restrict__ W1t,
                      const float* __restrict__ b1, const float* __restrict__ W2,
                      const float* __restrict__ b2, float* __restrict__ out) {
  __shared__ __align__(16) unsigned char smem[SMEM_BYTES];
  unsigned short* xn = (unsigned short*)smem;       // normalized bf16 rows
  unsigned char*  wb = smem + XN_BYTES;             // W double buffer
  float* hbuf = (float*)smem;                       // reuses xn region later

  const int tid  = threadIdx.x;
  const int lane = tid & 31;
  const int wave = tid >> 5;
  const long long r0 = (long long)blockIdx.x * ROWS;

  // ---- prefetch W chunk 0 while LayerNorm runs ----
#if USE_TDM
  if (wave == 0) tdm_issue(W1t, (unsigned)(unsigned long long)(uintptr_t)wb);
#else
  {
    const u32x4* src = (const u32x4*)(W1t + (size_t)tid * K_DIM);
    u32x4* dst = (u32x4*)(wb + tid * WB_ROW_B);
    dst[0] = src[0]; dst[1] = src[1]; dst[2] = src[2]; dst[3] = src[3];
  }
#endif

  // ---- LayerNorm (fp32): wave w handles rows 8w..8w+7, one wave per row ----
  for (int r = wave * 8; r < wave * 8 + 8; ++r) {
    const float4* xr = (const float4*)(X + (r0 + r) * K_DIM);
    float4 v[8];
    float s = 0.f, s2 = 0.f;
#pragma unroll
    for (int i = 0; i < 8; ++i) {
      v[i] = xr[lane + i * 32];
      s  += v[i].x + v[i].y + v[i].z + v[i].w;
      s2 += v[i].x * v[i].x + v[i].y * v[i].y + v[i].z * v[i].z + v[i].w * v[i].w;
    }
#pragma unroll
    for (int off = 16; off >= 1; off >>= 1) {
      s  += __shfl_xor(s, off, 32);
      s2 += __shfl_xor(s2, off, 32);
    }
    const float mu   = s * (1.f / K_DIM);
    const float rstd = rsqrtf(s2 * (1.f / K_DIM) - mu * mu + 1e-5f);
    unsigned short* dst = xn + r * XN_STRIDE_E;
#pragma unroll
    for (int i = 0; i < 8; ++i) {
      const int c = lane * 4 + i * 128;
      const float4 g = *(const float4*)(lnw + c);
      const float4 b = *(const float4*)(lnb + c);
      unsigned long long p =
            (unsigned long long)f2bf((v[i].x - mu) * rstd * g.x + b.x)
          | ((unsigned long long)f2bf((v[i].y - mu) * rstd * g.y + b.y) << 16)
          | ((unsigned long long)f2bf((v[i].z - mu) * rstd * g.z + b.z) << 32)
          | ((unsigned long long)f2bf((v[i].w - mu) * rstd * g.w + b.w) << 48);
      *(unsigned long long*)(dst + c) = p;
    }
  }

#if USE_TDM
  if (wave == 0) __builtin_amdgcn_s_wait_tensorcnt(0);
#endif
  __syncthreads();

  // ---- GEMM1: (64x1024 bf16) @ (1024x256 bf16) -> fp32, double-buffered K ----
  const int rt    = wave & 3;          // row tile (16 rows)
  const int nbase = (wave >> 2) * 128; // N half
  f32x8 acc[8];
#pragma unroll
  for (int t = 0; t < 8; ++t)
#pragma unroll
    for (int j = 0; j < 8; ++j) acc[t][j] = 0.f;

  const unsigned char* abase = (const unsigned char*)xn
      + (rt * 16 + (lane & 15)) * (XN_STRIDE_E * 2) + ((lane >> 4) * 16);
  const int bofs = (lane >> 4) * 16;

  for (int c = 0; c < NCHUNKS; ++c) {
    const unsigned char* wcur = wb + (c & 1) * WB_BYTES;
#if USE_TDM
    if (wave == 0 && c + 1 < NCHUNKS)
      tdm_issue(W1t + (c + 1) * KSTEP,
                (unsigned)(unsigned long long)(uintptr_t)(wb + ((c + 1) & 1) * WB_BYTES));
#else
    if (c + 1 < NCHUNKS) {
      const u32x4* src = (const u32x4*)(W1t + (size_t)tid * K_DIM + (c + 1) * KSTEP);
      u32x4* dst = (u32x4*)(wb + ((c + 1) & 1) * WB_BYTES + tid * WB_ROW_B);
      dst[0] = src[0]; dst[1] = src[1]; dst[2] = src[2]; dst[3] = src[3];
    }
#endif
    Frag A;
    A.q[0] = *(const u32x4*)(abase + c * 64);
    A.q[1] = *(const u32x4*)(abase + c * 64 + 32);
#pragma unroll
    for (int t = 0; t < 8; ++t) {
      const int n = nbase + t * 16 + (lane & 15);
      const unsigned char* bp = wcur + n * WB_ROW_B + bofs;
      Frag B;
      B.q[0] = *(const u32x4*)bp;
      B.q[1] = *(const u32x4*)(bp + 32);
      acc[t] = __builtin_amdgcn_wmma_f32_16x16x32_bf16(
          false, A.v, false, B.v, (short)0, acc[t], false, false);
    }
#if USE_TDM
    if (wave == 0 && c + 1 < NCHUNKS) __builtin_amdgcn_s_wait_tensorcnt(0);
#endif
    __syncthreads();
  }

  // ---- +b1, exact GELU, spill H (fp32) to LDS (xn region now free) ----
#pragma unroll
  for (int t = 0; t < 8; ++t) {
    const int col = nbase + t * 16 + (lane & 15);
    const float bv = b1[col];
#pragma unroll
    for (int j = 0; j < 8; ++j) {
      const int row = rt * 16 + ((lane >> 4) << 3) + j;
      float h = acc[t][j] + bv;
      h = 0.5f * h * (1.0f + erff(h * 0.70710678118654752f));
      hbuf[row * H_STRIDE + col] = h;
    }
  }
  __syncthreads();

  // ---- GEMM2: (64x256) @ (256x2) + b2, trivial VALU dot products ----
  if (tid < ROWS * 2) {
    const int row = tid >> 1;
    const int o = tid & 1;
    const float* h = hbuf + row * H_STRIDE;
    float a = 0.f;
#pragma unroll 4
    for (int n = 0; n < N_DIM; ++n) a += h[n] * W2[n * 2 + o];
    out[(r0 + row) * 2 + o] = a + b2[o];
  }
}

extern "C" void kernel_launch(void* const* d_in, const int* in_sizes, int n_in,
                              void* d_out, int out_size, void* d_ws, size_t ws_size,
                              hipStream_t stream) {
  const float* emb = (const float*)d_in[0];
  const float* lnw = (const float*)d_in[1];
  const float* lnb = (const float*)d_in[2];
  const float* W1  = (const float*)d_in[3];
  const float* b1  = (const float*)d_in[4];
  const float* W2  = (const float*)d_in[5];
  const float* b2  = (const float*)d_in[6];
  float* out = (float*)d_out;
  unsigned short* W1t = (unsigned short*)d_ws;  // 512 KB bf16 transposed W1

  prep_w1_kernel<<<(K_DIM * N_DIM) / 256, 256, 0, stream>>>(W1, W1t);
  head_main_kernel<<<BATCH / ROWS, THREADS, 0, stream>>>(emb, lnw, lnb, W1t, b1,
                                                         W2, b2, out);
}
